// QLinearLayer_68238440399495
// MI455X (gfx1250) — compile-verified, matrix operand order checked by reading, not run on hardware
//
#include <hip/hip_runtime.h>
#include <hip/hip_bf16.h>

typedef __attribute__((ext_vector_type(16))) _Float16 v16h;
typedef __attribute__((ext_vector_type(8)))  _Float16 v8h;
typedef __attribute__((ext_vector_type(8)))  float    v8f;
typedef __attribute__((ext_vector_type(4)))  int      v4i;

#define KDIM 4096
#define BM 128
#define BN 128
#define BK 32
#define LDA 40   // padded LDS row stride in halves (20 dwords -> conflict-free b128)

#if defined(__has_builtin)
#if __has_builtin(__builtin_amdgcn_global_load_async_to_lds_b128)
#define HAVE_ASYNC_LDS 1
#endif
#endif
#ifndef HAVE_ASYNC_LDS
#define HAVE_ASYNC_LDS 0
#endif

// ---------------------------------------------------------------------------
// Pass 1: dequantize int32 quantized values (int4/int6/int8 ranges) with
// per-32-group scales into a packed f16 matrix [rows, KDIM] at column offset.
// Read-once inputs -> non-temporal loads (keep L2 for the GEMM's B matrix).
// ---------------------------------------------------------------------------
__global__ __launch_bounds__(256) void dequant_kernel(
    const int* __restrict__ Q, const float* __restrict__ SF,
    _Float16* __restrict__ out, int p, int colOff, long total8)
{
    long idx = (long)blockIdx.x * blockDim.x + threadIdx.x;
    if (idx >= total8) return;
    int perRow = p >> 3;                       // 8 elements per thread
    int row = (int)(idx / perRow);
    int col = ((int)(idx - (long)row * perRow)) << 3;
    float sf = SF[(long)row * (p >> 5) + (col >> 5)];
    const v4i* q = (const v4i*)(Q + (long)row * p + col);
    v4i q0 = __builtin_nontemporal_load(q);
    v4i q1 = __builtin_nontemporal_load(q + 1);
    v8h h;
    h[0] = (_Float16)((float)q0[0] * sf);
    h[1] = (_Float16)((float)q0[1] * sf);
    h[2] = (_Float16)((float)q0[2] * sf);
    h[3] = (_Float16)((float)q0[3] * sf);
    h[4] = (_Float16)((float)q1[0] * sf);
    h[5] = (_Float16)((float)q1[1] * sf);
    h[6] = (_Float16)((float)q1[2] * sf);
    h[7] = (_Float16)((float)q1[3] * sf);
    *(v8h*)(out + (long)row * KDIM + colOff + col) = h;
}

// ---------------------------------------------------------------------------
// Fragment load from LDS per CDNA5 16-bit A/B layout (ISA 7.12.2):
// lane L (0-15): row L, K = kb..kb+7 and kb+16..kb+23  (kb = (lane>=16)?8:0)
// lane L+16   : row L, K = 8..15 and 24..31
// Two ds_load_b128 per fragment; LDA padding keeps them bank-conflict free.
// ---------------------------------------------------------------------------
__device__ __forceinline__ v16h frag_ld(const _Float16* tile, int row, int kb)
{
    const _Float16* p = tile + row * LDA + kb;
    v16h f;
    uint4* fp = (uint4*)&f;
    fp[0] = *(const uint4*)(p);        // K = kb .. kb+7
    fp[1] = *(const uint4*)(p + 16);   // K = kb+16 .. kb+23
    return f;
}

// ---------------------------------------------------------------------------
// Pass 2: f16 WMMA GEMM: out[M,N] = Aw[M,K] * Bw[N,K]^T + bias
// 128x128 tile / 256 threads (8 waves). Each wave: 32x64 sub-tile =
// 2x4 fragments = 8 v_wmma_f32_16x16x32_f16 per K-step of 32.
// Async path: 3-stage LDS pipeline fed by global_load_async_to_lds_b128
// (ASYNCcnt), prefetch distance 2. Fallback: 2-stage VGPR-staged pipeline.
// ---------------------------------------------------------------------------
__global__ __launch_bounds__(256) void gemm_kernel(
    const _Float16* __restrict__ Aw, const _Float16* __restrict__ Bw,
    const float* __restrict__ bias, float* __restrict__ out, int M, int N)
{
#if HAVE_ASYNC_LDS
    __shared__ _Float16 ldsA[3][BM * LDA];
    __shared__ _Float16 ldsB[3][BN * LDA];
#else
    __shared__ _Float16 ldsA[2][BM * LDA];
    __shared__ _Float16 ldsB[2][BN * LDA];
#endif

    const int tid  = threadIdx.x;
    const int lane = tid & 31;
    const int wave = tid >> 5;
    const int wm = wave & 3;      // 4 M sub-blocks of 32 rows
    const int wn = wave >> 2;     // 2 N sub-blocks of 64 cols
    const int lr = lane & 15;
    const int lh = lane >> 4;     // which half of the wave
    const int kb = lh * 8;        // fragment K base per lane-half

    const int tileM = blockIdx.y * BM;
    const int tileN = blockIdx.x * BN;

    // global->LDS staging: 2 threads per row, 16 halves (32B) each
    const int lrow = tid >> 1;
    const int lcol = (tid & 1) * 16;
    const _Float16* gA = Aw + (size_t)(tileM + lrow) * KDIM + lcol;
    const _Float16* gB = Bw + (size_t)(tileN + lrow) * KDIM + lcol;

    v8f acc[2][4] = {};
    const int NK = KDIM / BK;   // 128 K-chunks

#if HAVE_ASYNC_LDS
    typedef __attribute__((address_space(1))) v4i gv4i;   // global b128 unit
    typedef __attribute__((address_space(3))) v4i lv4i;   // LDS b128 unit

    // one tile copy = 4 async b128 ops per thread (2 for A, 2 for B)
    auto issue_tile = [&](int kt, int buf) {
        const _Float16* ga = gA + (size_t)kt * BK;
        const _Float16* gb = gB + (size_t)kt * BK;
        _Float16* sA = &ldsA[buf][lrow * LDA + lcol];
        _Float16* sB = &ldsB[buf][lrow * LDA + lcol];
        __builtin_amdgcn_global_load_async_to_lds_b128(
            (gv4i*)ga,       (lv4i*)sA,       0, 0);
        __builtin_amdgcn_global_load_async_to_lds_b128(
            (gv4i*)(ga + 8), (lv4i*)(sA + 8), 0, 0);
        __builtin_amdgcn_global_load_async_to_lds_b128(
            (gv4i*)gb,       (lv4i*)sB,       0, 0);
        __builtin_amdgcn_global_load_async_to_lds_b128(
            (gv4i*)(gb + 8), (lv4i*)(sB + 8), 0, 0);
    };

    issue_tile(0, 0);
    issue_tile(1, 1);

    for (int it = 0; it < NK; ++it) {
        // complete the oldest in-flight tile copy (4 ops), keep the next going
        if (it + 1 < NK) asm volatile("s_wait_asynccnt %0" :: "n"(4) : "memory");
        else             asm volatile("s_wait_asynccnt %0" :: "n"(0) : "memory");
        __syncthreads();
        if (it + 2 < NK) issue_tile(it + 2, (it + 2) % 3);

        const int cur = it % 3;
        v16h af[2], bf[4];
        #pragma unroll
        for (int m = 0; m < 2; ++m)
            af[m] = frag_ld(ldsA[cur], wm * 32 + m * 16 + lr, kb);
        #pragma unroll
        for (int n = 0; n < 4; ++n)
            bf[n] = frag_ld(ldsB[cur], wn * 64 + n * 16 + lr, kb);

        #pragma unroll
        for (int m = 0; m < 2; ++m)
            #pragma unroll
            for (int n = 0; n < 4; ++n)
                acc[m][n] = __builtin_amdgcn_wmma_f32_16x16x32_f16(
                    false, af[m], false, bf[n], (short)0, acc[m][n], false, false);
    }
#else
    uint4 ra[2], rb[2];
    // prologue: fetch K-chunk 0 and stage into buffer 0
    #pragma unroll
    for (int j = 0; j < 2; ++j) {
        ra[j] = *(const uint4*)(gA + j * 8);
        rb[j] = *(const uint4*)(gB + j * 8);
    }
    {
        _Float16* sA = &ldsA[0][lrow * LDA + lcol];
        _Float16* sB = &ldsB[0][lrow * LDA + lcol];
        #pragma unroll
        for (int j = 0; j < 2; ++j) {
            *(uint4*)(sA + j * 8) = ra[j];
            *(uint4*)(sB + j * 8) = rb[j];
        }
    }

    for (int it = 0; it < NK; ++it) {
        __syncthreads();
        const int cur  = it & 1;
        const bool more = (it + 1) < NK;
        if (more) {
            const _Float16* ga = gA + (size_t)(it + 1) * BK;
            const _Float16* gb = gB + (size_t)(it + 1) * BK;
            #pragma unroll
            for (int j = 0; j < 2; ++j) {
                ra[j] = *(const uint4*)(ga + j * 8);
                rb[j] = *(const uint4*)(gb + j * 8);
            }
        }

        v16h af[2], bf[4];
        #pragma unroll
        for (int m = 0; m < 2; ++m)
            af[m] = frag_ld(ldsA[cur], wm * 32 + m * 16 + lr, kb);
        #pragma unroll
        for (int n = 0; n < 4; ++n)
            bf[n] = frag_ld(ldsB[cur], wn * 64 + n * 16 + lr, kb);

        #pragma unroll
        for (int m = 0; m < 2; ++m)
            #pragma unroll
            for (int n = 0; n < 4; ++n)
                acc[m][n] = __builtin_amdgcn_wmma_f32_16x16x32_f16(
                    false, af[m], false, bf[n], (short)0, acc[m][n], false, false);

        if (more) {
            const int nxt = cur ^ 1;
            _Float16* sA = &ldsA[nxt][lrow * LDA + lcol];
            _Float16* sB = &ldsB[nxt][lrow * LDA + lcol];
            #pragma unroll
            for (int j = 0; j < 2; ++j) {
                *(uint4*)(sA + j * 8) = ra[j];
                *(uint4*)(sB + j * 8) = rb[j];
            }
        }
    }
#endif

    // epilogue: C/D layout -> VGPR r holds M=r (lanes 0-15) / M=r+8 (16-31).
    // Output is write-once/stream -> non-temporal stores keep L2 for B.
    #pragma unroll
    for (int n = 0; n < 4; ++n) {
        const int gn = tileN + wn * 64 + n * 16 + lr;
        const float bv = bias[gn];
        #pragma unroll
        for (int m = 0; m < 2; ++m) {
            const int gm0 = tileM + wm * 32 + m * 16 + lh * 8;
            #pragma unroll
            for (int r = 0; r < 8; ++r)
                __builtin_nontemporal_store(acc[m][n][r] + bv,
                                            &out[(size_t)(gm0 + r) * N + gn]);
        }
    }
}

// ---------------------------------------------------------------------------
extern "C" void kernel_launch(void* const* d_in, const int* in_sizes, int n_in,
                              void* d_out, int out_size, void* d_ws, size_t ws_size,
                              hipStream_t stream)
{
    (void)n_in; (void)out_size; (void)ws_size;
    const int*   AN   = (const int*)d_in[0];
    const int*   AS   = (const int*)d_in[1];
    const int*   AO   = (const int*)d_in[2];
    const float* SFAN = (const float*)d_in[3];
    const float* SFAS = (const float*)d_in[4];
    const float* SFAO = (const float*)d_in[5];
    const int*   BN_  = (const int*)d_in[6];
    const int*   BS   = (const int*)d_in[7];
    const int*   BO   = (const int*)d_in[8];
    const float* SFBN = (const float*)d_in[9];
    const float* SFBS = (const float*)d_in[10];
    const float* SFBO = (const float*)d_in[11];
    const float* bias = (const float*)d_in[12];

    const int M = in_sizes[0] / 2048;   // 8192
    const int N = in_sizes[6] / 2048;   // 4096

    _Float16* Aw = (_Float16*)d_ws;                 // M x KDIM f16  (64 MB)
    _Float16* Bw = Aw + (size_t)M * KDIM;           // N x KDIM f16  (32 MB)

    auto dq = [&](const int* Q, const float* SF, _Float16* outp,
                  int rows, int p, int colOff) {
        long total8 = (long)rows * p / 8;
        int blocks = (int)((total8 + 255) / 256);
        dequant_kernel<<<blocks, 256, 0, stream>>>(Q, SF, outp, p, colOff, total8);
    };

    dq(AN,  SFAN, Aw, M, 2048, 0);
    dq(AS,  SFAS, Aw, M, 1024, 2048);
    dq(AO,  SFAO, Aw, M, 1024, 3072);
    dq(BN_, SFBN, Bw, N, 2048, 0);
    dq(BS,  SFBS, Bw, N, 1024, 2048);
    dq(BO,  SFBO, Bw, N, 1024, 3072);

    dim3 grid(N / BN, M / BM);
    gemm_kernel<<<grid, 256, 0, stream>>>(Aw, Bw, bias, (float*)d_out, M, N);
}
